// Multihead_Attention_59081570123872
// MI455X (gfx1250) — compile-verified, matrix operand order checked by reading, not run on hardware
//
#include <hip/hip_runtime.h>

#define D_MODEL 1024
#define N_HEADS 16
#define D_K     64
#define SEQ     2048
#define BATCH   4
#define ROWS    (BATCH * SEQ)   // 8192

typedef __attribute__((ext_vector_type(16))) _Float16 v16h;
typedef __attribute__((ext_vector_type(8)))  _Float16 v8h;
typedef __attribute__((ext_vector_type(8)))  float    v8f;
typedef __attribute__((ext_vector_type(4)))  float    v4f;
typedef int i32x4 __attribute__((vector_size(4 * sizeof(int))));

#ifndef __has_builtin
#define __has_builtin(x) 0
#endif
#if __has_builtin(__builtin_amdgcn_global_load_async_to_lds_b128)
#define HAVE_ASYNC_LDS 1
#else
#define HAVE_ASYNC_LDS 0
#endif

// ---------------------------------------------------------------------------
// WMMA A/B fragment loader (CDNA5 16-bit 16x32 layout, wave32):
//   lanes 0-15 : row = lane,    K = {0..7, 16..23}
//   lanes 16-31: row = lane-16, K = {8..15, 24..31}
// Caller passes pointer already offset to (row, kc + hi*8).
// ---------------------------------------------------------------------------
__device__ inline v16h frag_from_f16(const _Float16* p) {
  v8h lo = *(const v8h*)(p);
  v8h hh = *(const v8h*)(p + 16);
  v16h a;
#pragma unroll
  for (int i = 0; i < 8; ++i) { a[i] = lo[i]; a[i + 8] = hh[i]; }
  return a;
}

// Row reductions across the 16 lanes holding one C-matrix row (N striping).
__device__ inline float rowmax16(float v) {
#pragma unroll
  for (int m = 1; m < 16; m <<= 1) v = fmaxf(v, __shfl_xor(v, m, 32));
  return v;
}
__device__ inline float rowsum16(float v) {
#pragma unroll
  for (int m = 1; m < 16; m <<= 1) v += __shfl_xor(v, m, 32);
  return v;
}

// 16-byte global -> LDS copy: async gfx1250 path (ASYNCcnt-tracked) if the
// toolchain exposes the builtin, otherwise a synchronous bounce through VGPRs.
__device__ inline void copy16_to_lds(const _Float16* gsrc, _Float16* ldst) {
#if HAVE_ASYNC_LDS
  i32x4* gs = (i32x4*)const_cast<_Float16*>(gsrc);
  i32x4* ls = (i32x4*)ldst;
  __builtin_amdgcn_global_load_async_to_lds_b128(
      (__attribute__((address_space(1))) i32x4*)gs,
      (__attribute__((address_space(3))) i32x4*)ls, 0, 0);
#else
  *(v8h*)ldst = *(const v8h*)gsrc;
#endif
}

__device__ inline void wait_async_copies() {
#if HAVE_ASYNC_LDS
#if __has_builtin(__builtin_amdgcn_s_wait_asynccnt)
  __builtin_amdgcn_s_wait_asynccnt(0);
#else
  asm volatile("s_wait_asynccnt 0" ::: "memory");
#endif
#endif
}

// ---------------------------------------------------------------------------
// Kernel 0: bulk fp32 -> f16 conversion (vectorized, 8 elements/thread).
// ---------------------------------------------------------------------------
__global__ __launch_bounds__(256)
void cvt_f16_kernel(const float* __restrict__ src, _Float16* __restrict__ dst,
                    int n8) {
  const int i = blockIdx.x * 256 + threadIdx.x;
  if (i < n8) {
    const v4f* s = (const v4f*)src + (size_t)i * 2;
    v4f x0 = s[0], x1 = s[1];
    v8h o;
#pragma unroll
    for (int k = 0; k < 4; ++k) {
      o[k]     = (_Float16)x0[k];
      o[k + 4] = (_Float16)x1[k];
    }
    ((v8h*)dst)[i] = o;
  }
}

// ---------------------------------------------------------------------------
// Kernel 1: Q/K/V = x @ W^T (f16 operands), fused RoPE on Q/K.
// Wave tile 16(M) x 64(N): A frag reused across 4 WMMAs per K-step.
//   Q,K stored f16 [b, h, s, d];  V stored transposed f16 [b, h, d, s].
// grid = (ROWS/16, 2, 3), block = 256 (8 waves)
// ---------------------------------------------------------------------------
__global__ __launch_bounds__(256)
void qkv_rope_kernel(const _Float16* __restrict__ xh,
                     const int*      __restrict__ tpos,
                     const _Float16* __restrict__ Wqh,
                     const _Float16* __restrict__ Wkh,
                     const _Float16* __restrict__ Wvh,
                     _Float16* __restrict__ Qh,
                     _Float16* __restrict__ Kh,
                     _Float16* __restrict__ Vt) {
  const int lane = threadIdx.x & 31;
  const int wave = threadIdx.x >> 5;
  const int hi   = (lane >= 16) ? 1 : 0;
  const int lo16 = lane & 15;

  const int m0 = blockIdx.x * 16;
  const int n0 = (blockIdx.y * 8 + wave) * 64;
  const int z  = blockIdx.z;  // 0=Q 1=K 2=V
  const _Float16* W = (z == 0) ? Wqh : (z == 1) ? Wkh : Wvh;

  const size_t arow = (size_t)(m0 + lo16) * D_MODEL;

  v8f acc[4];
#pragma unroll
  for (int j = 0; j < 4; ++j)
#pragma unroll
    for (int r = 0; r < 8; ++r) acc[j][r] = 0.0f;

#pragma unroll 2
  for (int kc = 0; kc < D_MODEL; kc += 32) {
    v16h a = frag_from_f16(xh + arow + kc + hi * 8);
#pragma unroll
    for (int j = 0; j < 4; ++j) {
      v16h b = frag_from_f16(W + (size_t)(n0 + j * 16 + lo16) * D_MODEL +
                             kc + hi * 8);
      acc[j] = __builtin_amdgcn_wmma_f32_16x16x32_f16(false, a, false, b,
                                                      (short)0, acc[j],
                                                      false, false);
    }
  }

#pragma unroll
  for (int j = 0; j < 4; ++j) {
    const int n = n0 + j * 16 + lo16;
    const int h = n >> 6;
    const int d = n & 63;

    if (z < 2) {
      // RoPE: even/odd features sit in adjacent lanes of the C fragment.
      const int   pairIdx = d >> 1;
      const float inv = __powf(10000.0f, -(float)(2 * pairIdx) / (float)D_K);
      const bool  isOdd = (d & 1) != 0;
      _Float16* dst = (z == 0) ? Qh : Kh;
#pragma unroll
      for (int r = 0; r < 8; ++r) {
        const int m = m0 + r + hi * 8;
        const int s = m & (SEQ - 1);
        const int b = m >> 11;
        const float pos = (float)tpos[s];
        float sn, cs;
        __sincosf(pos * inv, &sn, &cs);
        float mine  = acc[j][r];
        float other = __shfl_xor(mine, 1, 32);
        float out   = isOdd ? (other * sn + mine * cs)
                            : (mine * cs - other * sn);
        dst[(((size_t)(b * N_HEADS + h) * SEQ) + s) * D_K + d] = (_Float16)out;
      }
    } else {
#pragma unroll
      for (int r = 0; r < 8; ++r) {
        const int m = m0 + r + hi * 8;
        const int s = m & (SEQ - 1);
        const int b = m >> 11;
        Vt[(((size_t)(b * N_HEADS + h) * D_K) + d) * SEQ + s] =
            (_Float16)acc[j][r];
      }
    }
  }
}

// ---------------------------------------------------------------------------
// Kernel 2: causal flash attention.
// Block = 4 waves handling 64 consecutive queries of one (b,h); K/V tiles
// (32 keys) staged into double-buffered LDS with async global->LDS copies and
// shared by all 4 waves; per-wave P-tile transposed through LDS.
// grid = (SEQ/64, BATCH*N_HEADS), block = 128
// ---------------------------------------------------------------------------
__device__ inline void stage_kv_tiles(const _Float16* Kb, const _Float16* Vb,
                                      int kt0, _Float16* Kt, _Float16* Vl,
                                      int tid) {
  // K tile: 32 rows x 64 halfs (row stride 72) = 256 x 16B chunks
#pragma unroll
  for (int i = 0; i < 2; ++i) {
    const int q   = tid + i * 128;
    const int row = q >> 3, c = q & 7;
    copy16_to_lds(Kb + (size_t)(kt0 + row) * D_K + c * 8,
                  Kt + row * 72 + c * 8);
  }
  // V^T tile: 64 rows (d) x 32 halfs (keys), row stride 40
#pragma unroll
  for (int i = 0; i < 2; ++i) {
    const int q   = tid + i * 128;
    const int row = q >> 2, c = q & 3;
    copy16_to_lds(Vb + (size_t)row * SEQ + kt0 + c * 8,
                  Vl + row * 40 + c * 8);
  }
}

__global__ __launch_bounds__(128)
void flash_attn_kernel(const _Float16* __restrict__ Qh,
                       const _Float16* __restrict__ Kh,
                       const _Float16* __restrict__ Vt,
                       _Float16* __restrict__ ctx) {
  __shared__ __align__(16) _Float16 Ktile[2][32 * 72];
  __shared__ __align__(16) _Float16 Vtile[2][64 * 40];
  __shared__ __align__(16) _Float16 pbuf[4][16 * 40];

  const int tid  = threadIdx.x;
  const int lane = tid & 31;
  const int wave = tid >> 5;
  const int hi   = (lane >= 16) ? 1 : 0;
  const int lo16 = lane & 15;

  const int bh = blockIdx.y;
  const int Q0 = blockIdx.x * 64;
  const int q0 = Q0 + wave * 16;

  const _Float16* Qb = Qh + (size_t)bh * SEQ * D_K;
  const _Float16* Kb = Kh + (size_t)bh * SEQ * D_K;
  const _Float16* Vb = Vt + (size_t)bh * D_K * SEQ;
  _Float16* pb = pbuf[wave];

  // Q fragments (16 queries x 64 dims), kept in registers.
  const v16h aq0 = frag_from_f16(Qb + (size_t)(q0 + lo16) * D_K + 0  + hi * 8);
  const v16h aq1 = frag_from_f16(Qb + (size_t)(q0 + lo16) * D_K + 32 + hi * 8);

  v8f acc[4];
  float mrow[8], lrow[8];
#pragma unroll
  for (int t = 0; t < 4; ++t)
#pragma unroll
    for (int r = 0; r < 8; ++r) acc[t][r] = 0.0f;
#pragma unroll
  for (int r = 0; r < 8; ++r) { mrow[r] = -3.0e38f; lrow[r] = 0.0f; }

  const float scale = 0.125f;              // 1/sqrt(D_K)
  const int   nt    = blockIdx.x * 2 + 2;  // ceil((Q0+64)/32)

  stage_kv_tiles(Kb, Vb, 0, Ktile[0], Vtile[0], tid);

  for (int t = 0; t < nt; ++t) {
    const int kt0 = t * 32;
    wait_async_copies();
    __syncthreads();
    if (t + 1 < nt)
      stage_kv_tiles(Kb, Vb, (t + 1) * 32, Ktile[(t + 1) & 1],
                     Vtile[(t + 1) & 1], tid);

    if (kt0 <= q0 + 15) {   // causal: this wave still has unmasked keys
      const _Float16* Kt = Ktile[t & 1];
      const _Float16* Vl = Vtile[t & 1];

      // ---- scores: two 16-key subtiles, 2 WMMAs each (d_k = 64) ----
      v8f s0, s1;
#pragma unroll
      for (int r = 0; r < 8; ++r) { s0[r] = 0.0f; s1[r] = 0.0f; }
      {
        v16h b0 = frag_from_f16(Kt + (lo16) * 72 + 0  + hi * 8);
        v16h b1 = frag_from_f16(Kt + (lo16) * 72 + 32 + hi * 8);
        s0 = __builtin_amdgcn_wmma_f32_16x16x32_f16(false, aq0, false, b0,
                                                    (short)0, s0, false, false);
        s0 = __builtin_amdgcn_wmma_f32_16x16x32_f16(false, aq1, false, b1,
                                                    (short)0, s0, false, false);
      }
      {
        v16h b0 = frag_from_f16(Kt + (16 + lo16) * 72 + 0  + hi * 8);
        v16h b1 = frag_from_f16(Kt + (16 + lo16) * 72 + 32 + hi * 8);
        s1 = __builtin_amdgcn_wmma_f32_16x16x32_f16(false, aq0, false, b0,
                                                    (short)0, s1, false, false);
        s1 = __builtin_amdgcn_wmma_f32_16x16x32_f16(false, aq1, false, b1,
                                                    (short)0, s1, false, false);
      }

      // ---- scale + causal mask + online softmax ----
      float alpha[8];
#pragma unroll
      for (int r = 0; r < 8; ++r) {
        const int q   = q0 + r + hi * 8;
        const int k0i = kt0 + lo16;
        const int k1i = kt0 + 16 + lo16;
        s0[r] = (k0i <= q) ? s0[r] * scale : -3.0e38f;
        s1[r] = (k1i <= q) ? s1[r] * scale : -3.0e38f;

        float tmax = rowmax16(fmaxf(s0[r], s1[r]));
        float mnew = fmaxf(mrow[r], tmax);
        alpha[r]   = __expf(mrow[r] - mnew);
        mrow[r]    = mnew;
        float p0 = __expf(s0[r] - mnew);
        float p1 = __expf(s1[r] - mnew);
        lrow[r]  = lrow[r] * alpha[r] + rowsum16(p0 + p1);

        const int M = r + hi * 8;       // C layout -> LDS for transpose
        pb[M * 40 + lo16]      = (_Float16)p0;
        pb[M * 40 + 16 + lo16] = (_Float16)p1;
      }

#pragma unroll
      for (int tt = 0; tt < 4; ++tt)
#pragma unroll
        for (int r = 0; r < 8; ++r) acc[tt][r] *= alpha[r];

      asm volatile("s_wait_dscnt 0" ::: "memory");

      // P back in A layout (lane = query row, K = 32 keys)
      const v16h ap = frag_from_f16(&pb[lo16 * 40 + hi * 8]);

      // ---- acc += P @ V ----
#pragma unroll
      for (int tt = 0; tt < 4; ++tt) {
        v16h bv = frag_from_f16(Vl + (tt * 16 + lo16) * 40 + hi * 8);
        acc[tt] = __builtin_amdgcn_wmma_f32_16x16x32_f16(false, ap, false, bv,
                                                         (short)0, acc[tt],
                                                         false, false);
      }
    }
  }

  // ---- epilogue: normalize, scatter to [b, s, h*64+d] context (f16) ----
  const int b = bh >> 4;
  const int h = bh & 15;
#pragma unroll
  for (int tt = 0; tt < 4; ++tt) {
    const int e = h * 64 + tt * 16 + lo16;
#pragma unroll
    for (int r = 0; r < 8; ++r) {
      const int q = q0 + r + hi * 8;
      float o = acc[tt][r] / lrow[r];
      ctx[((size_t)(b * SEQ + q)) * D_MODEL + e] = (_Float16)o;
    }
  }
}

// ---------------------------------------------------------------------------
// Kernel 3: out = ctx @ Wo^T (all-f16 operands, f32 out), wave tile 16x64.
// grid = (ROWS/16, 2), block = 256
// ---------------------------------------------------------------------------
__global__ __launch_bounds__(256)
void out_proj_kernel(const _Float16* __restrict__ ctx,
                     const _Float16* __restrict__ Woh,
                     float* __restrict__ out) {
  const int lane = threadIdx.x & 31;
  const int wave = threadIdx.x >> 5;
  const int hi   = (lane >= 16) ? 1 : 0;
  const int lo16 = lane & 15;

  const int m0 = blockIdx.x * 16;
  const int n0 = (blockIdx.y * 8 + wave) * 64;
  const size_t arow = (size_t)(m0 + lo16) * D_MODEL;

  v8f acc[4];
#pragma unroll
  for (int j = 0; j < 4; ++j)
#pragma unroll
    for (int r = 0; r < 8; ++r) acc[j][r] = 0.0f;

#pragma unroll 2
  for (int kc = 0; kc < D_MODEL; kc += 32) {
    v16h a = frag_from_f16(ctx + arow + kc + hi * 8);
#pragma unroll
    for (int j = 0; j < 4; ++j) {
      v16h b = frag_from_f16(Woh + (size_t)(n0 + j * 16 + lo16) * D_MODEL +
                             kc + hi * 8);
      acc[j] = __builtin_amdgcn_wmma_f32_16x16x32_f16(false, a, false, b,
                                                      (short)0, acc[j],
                                                      false, false);
    }
  }

#pragma unroll
  for (int j = 0; j < 4; ++j)
#pragma unroll
    for (int r = 0; r < 8; ++r) {
      const int m = m0 + r + hi * 8;
      out[(size_t)m * D_MODEL + n0 + j * 16 + lo16] = acc[j][r];
    }
}

// ---------------------------------------------------------------------------
extern "C" void kernel_launch(void* const* d_in, const int* in_sizes, int n_in,
                              void* d_out, int out_size, void* d_ws,
                              size_t ws_size, hipStream_t stream) {
  (void)in_sizes; (void)n_in; (void)out_size; (void)ws_size;
  const float* x    = (const float*)d_in[0];
  const int*   tpos = (const int*)  d_in[1];
  const float* Wq   = (const float*)d_in[2];
  const float* Wk   = (const float*)d_in[3];
  const float* Wv   = (const float*)d_in[4];
  const float* Wo   = (const float*)d_in[5];
  float* out = (float*)d_out;

  char* ws = (char*)d_ws;
  const size_t seg  = (size_t)ROWS * D_MODEL * sizeof(_Float16);     // 16 MB
  const size_t wseg = (size_t)D_MODEL * D_MODEL * sizeof(_Float16);  //  2 MB
  _Float16* xh  = (_Float16*)(ws);            // aliased: ctx reuses this slot
  _Float16* ctx = (_Float16*)(ws);            // (xh dead after kernel 1)
  _Float16* Qh  = (_Float16*)(ws + seg);
  _Float16* Kh  = (_Float16*)(ws + 2 * seg);
  _Float16* Vt  = (_Float16*)(ws + 3 * seg);
  _Float16* Wqh = (_Float16*)(ws + 4 * seg);
  _Float16* Wkh = (_Float16*)(ws + 4 * seg + wseg);
  _Float16* Wvh = (_Float16*)(ws + 4 * seg + 2 * wseg);
  _Float16* Woh = (_Float16*)(ws + 4 * seg + 3 * wseg);

  // fp32 -> f16 staging (one pass; removes conversions from GEMM hot loops)
  const int nx8 = ROWS * D_MODEL / 8;
  const int nw8 = D_MODEL * D_MODEL / 8;
  cvt_f16_kernel<<<(nx8 + 255) / 256, 256, 0, stream>>>(x, xh, nx8);
  cvt_f16_kernel<<<(nw8 + 255) / 256, 256, 0, stream>>>(Wq, Wqh, nw8);
  cvt_f16_kernel<<<(nw8 + 255) / 256, 256, 0, stream>>>(Wk, Wkh, nw8);
  cvt_f16_kernel<<<(nw8 + 255) / 256, 256, 0, stream>>>(Wv, Wvh, nw8);
  cvt_f16_kernel<<<(nw8 + 255) / 256, 256, 0, stream>>>(Wo, Woh, nw8);

  dim3 g1(ROWS / 16, 2, 3);
  qkv_rope_kernel<<<g1, 256, 0, stream>>>(xh, tpos, Wqh, Wkh, Wvh, Qh, Kh, Vt);

  dim3 g2(SEQ / 64, BATCH * N_HEADS);
  flash_attn_kernel<<<g2, 128, 0, stream>>>(Qh, Kh, Vt, ctx);

  dim3 g3(ROWS / 16, 2);
  out_proj_kernel<<<g3, 256, 0, stream>>>(ctx, Woh, out);
}
